// MDLSTM_25434796327376
// MI455X (gfx1250) — matrix operand order repaired and checked by reading, not compiled, not run on hardware
//
#include <hip/hip_runtime.h>
#include <hip/hip_fp16.h>

#define H_ 128
#define W_ 128
#define B_ 8
#define K_ 128              // SIZE_IN
#define O_ 128              // SIZE_OUT
#define M_ 640              // 5 * O_  (pf0 | pf1 | pi | po | pc)
#define NMB 5               // number of 128-wide M blocks
#define N_ (H_ * W_ * B_)   // 131072 GEMM rows
#define TILE 16
#define TGRID (H_ / TILE)   // 8x8 tile grid
#define LDSK 136            // padded LDS row stride (f16 elems): avoids bank conflicts, keeps 16B align

typedef _Float16 v8h  __attribute__((ext_vector_type(8)));
typedef _Float16 v16h __attribute__((ext_vector_type(16)));
typedef float    v8f  __attribute__((ext_vector_type(8)));

// ---------------------------------------------------------------------------
// Pack all weight matrices into one transposed f16 matrix Wh[m][k] (M_ x K_)
// and all biases into biasAll[m].  Tiny; runs once per call.
// ---------------------------------------------------------------------------
__global__ void pack_kernel(const float* __restrict__ wf, const float* __restrict__ wi,
                            const float* __restrict__ wo, const float* __restrict__ wc,
                            const float* __restrict__ bf, const float* __restrict__ bi,
                            const float* __restrict__ bo, const float* __restrict__ bc,
                            _Float16* __restrict__ Wh, float* __restrict__ biasAll) {
  int m = blockIdx.x * blockDim.x + threadIdx.x;
  if (m >= M_) return;
  const float* src; int col; float bias;
  if (m < 256)      { src = wf + (size_t)(m >> 7) * K_ * O_; col = m & 127; bias = bf[m]; }
  else if (m < 384) { src = wi; col = m - 256; bias = bi[col]; }
  else if (m < 512) { src = wo; col = m - 384; bias = bo[col]; }
  else              { src = wc; col = m - 512; bias = bc[col]; }
  biasAll[m] = bias;
  for (int k = 0; k < K_; ++k)
    Wh[(size_t)m * K_ + k] = (_Float16)src[(size_t)k * O_ + col];
}

// ---------------------------------------------------------------------------
// P = X(131072x128) * W(128x640) + bias, via v_wmma_f32_16x16x32_f16.
// Block: 256 threads (8 wave32s).  One block owns a 16(N) x 640(M) strip:
// X staged to LDS ONCE (f32 -> f16), each wave holds 5 accumulators and
// computes 5 16x16 tiles, reusing each A fragment across all 5 M-blocks.
// X HBM traffic drops from 320 MB to 64 MB; Wh (160 KB) stays L2-resident.
// ---------------------------------------------------------------------------
__global__ void proj_gemm(const float* __restrict__ X, const _Float16* __restrict__ Wh,
                          const float* __restrict__ biasAll, float* __restrict__ P) {
  __shared__ _Float16 sA[16 * LDSK];
  const int tid  = threadIdx.x;
  const int lane = tid & 31, wave = tid >> 5;
  const int half = lane >> 4, ml = lane & 15;
  const int nBase = blockIdx.x * 16;

  // Stage the 16x128 f32 X strip into LDS as f16 (2048 elems / 256 threads = 8 each)
  {
    int row = tid >> 4;          // 0..15
    int k0  = (tid & 15) * 8;    // 0..120
    const float4* xp = (const float4*)(X + (size_t)(nBase + row) * K_ + k0);
    float4 x0 = xp[0], x1 = xp[1];
    _Float16* dst = sA + row * LDSK + k0;
    dst[0] = (_Float16)x0.x; dst[1] = (_Float16)x0.y;
    dst[2] = (_Float16)x0.z; dst[3] = (_Float16)x0.w;
    dst[4] = (_Float16)x1.x; dst[5] = (_Float16)x1.y;
    dst[6] = (_Float16)x1.z; dst[7] = (_Float16)x1.w;
  }
  __syncthreads();

  // 5 output tiles per wave: nCol[mb] = mb*128 + wave*16 + ml
  v8f acc[NMB];
#pragma unroll
  for (int mb = 0; mb < NMB; ++mb) {
    const float bias = biasAll[mb * 128 + wave * 16 + ml];
#pragma unroll
    for (int r = 0; r < 8; ++r) acc[mb][r] = bias;  // fold bias into accumulator init
  }

#pragma unroll
  for (int kc = 0; kc < 4; ++kc) {           // K = 4 chunks of 32
    // A fragment: row = ml; lane-half selects contiguous K subset {16h..16h+15}
    const _Float16* ap = sA + ml * LDSK + kc * 32 + half * 16;
    v8h a0 = *(const v8h*)ap;
    v8h a1 = *(const v8h*)(ap + 8);
    v16h a;
#pragma unroll
    for (int t = 0; t < 8; ++t) { a[t] = a0[t]; a[8 + t] = a1[t]; }
#pragma unroll
    for (int mb = 0; mb < NMB; ++mb) {
      const int nCol = mb * 128 + wave * 16 + ml;
      // B fragment: col = nCol, K = [32kc + 16*half, +16) -> one contiguous 32B load
      v16h b = *(const v16h*)(Wh + (size_t)nCol * K_ + kc * 32 + half * 16);
      acc[mb] = __builtin_amdgcn_wmma_f32_16x16x32_f16(false, a, false, b, (short)0,
                                                       acc[mb], false, false);
    }
  }

  // D layout: VGPR r -> (row = r + 8*half, col = lane&15)
#pragma unroll
  for (int mb = 0; mb < NMB; ++mb) {
    const int nCol = mb * 128 + wave * 16 + ml;
#pragma unroll
    for (int r = 0; r < 8; ++r) {
      int row = r + half * 8;
      P[(size_t)(nBase + row) * M_ + nCol] = acc[mb][r];
    }
  }
}

// ---------------------------------------------------------------------------
// 2D-LSTM wavefront recurrence over 16x16 cell tiles.  One 1024-thread block
// per tile (thread = one (b,o) element); 31 internal anti-diagonal steps with
// workgroup barriers.  Cross-tile deps satisfied by per-tile-diagonal launch.
// ---------------------------------------------------------------------------
__device__ __forceinline__ float sigf(float v) { return 1.0f / (1.0f + __expf(-v)); }

__global__ void mdlstm_tile(const float* __restrict__ P,
                            const float* __restrict__ uf, const float* __restrict__ ui,
                            const float* __restrict__ uo, const float* __restrict__ uc,
                            float* __restrict__ S, float* __restrict__ Hout,
                            int diag, int ti0) {
  const int TI = ti0 + blockIdx.x;
  const int TJ = diag - TI;
  const int b = threadIdx.x >> 7;    // 0..7
  const int o = threadIdx.x & 127;   // 0..127

  // Diagonal recurrent weights: fixed per (o), keep in registers
  const float uf00 = uf[o],       uf01 = uf[128 + o];
  const float uf10 = uf[256 + o], uf11 = uf[384 + o];
  const float ui0 = ui[o], ui1 = ui[128 + o];
  const float uo0 = uo[o], uo1 = uo[128 + o];
  const float uc0 = uc[o], uc1 = uc[128 + o];

  for (int t = 0; t < 2 * TILE - 1; ++t) {
    int clo = t - (TILE - 1); if (clo < 0) clo = 0;
    int chi = t < TILE - 1 ? t : TILE - 1;
    for (int ci = clo; ci <= chi; ++ci) {
      const int i = TI * TILE + ci;
      const int j = TJ * TILE + (t - ci);
      const size_t cell = (size_t)(i * W_ + j) * B_ + b;
      const size_t idx  = cell * O_ + o;
      const size_t pb   = cell * (size_t)M_;
      const float pf0 = P[pb + o],        pf1 = P[pb + 128 + o];
      const float pi  = P[pb + 256 + o],  po  = P[pb + 384 + o];
      const float pc  = P[pb + 512 + o];
      float hu = 0.f, su = 0.f, hl = 0.f, sl = 0.f;
      if (i) { hu = Hout[idx - (size_t)W_ * B_ * O_]; su = S[idx - (size_t)W_ * B_ * O_]; }
      if (j) { hl = Hout[idx - (size_t)B_ * O_];      sl = S[idx - (size_t)B_ * O_]; }
      const float f0 = sigf(pf0 + uf00 * hu + uf01 * hl);
      const float f1 = sigf(pf1 + uf10 * hu + uf11 * hl);
      const float ig = sigf(pi + ui0 * hu + ui1 * hl);
      const float og = sigf(po + uo0 * hu + uo1 * hl);
      const float cg = sigf(pc + uc0 * hu + uc1 * hl);
      const float s = ig * cg + f0 * su + f1 * sl;
      const float h = og * tanhf(s);
      S[idx] = s;
      Hout[idx] = h;
    }
    __threadfence_block();   // make this diagonal's global h/s visible in-block
    __syncthreads();
  }
}

// ---------------------------------------------------------------------------
extern "C" void kernel_launch(void* const* d_in, const int* in_sizes, int n_in,
                              void* d_out, int out_size, void* d_ws, size_t ws_size,
                              hipStream_t stream) {
  (void)in_sizes; (void)n_in; (void)out_size;
  const float* x  = (const float*)d_in[0];
  const float* wf = (const float*)d_in[1];
  const float* uf = (const float*)d_in[2];
  const float* bf = (const float*)d_in[3];
  const float* wi = (const float*)d_in[4];
  const float* ui = (const float*)d_in[5];
  const float* bi = (const float*)d_in[6];
  const float* wo = (const float*)d_in[7];
  const float* uo = (const float*)d_in[8];
  const float* bo = (const float*)d_in[9];
  const float* wc = (const float*)d_in[10];
  const float* uc = (const float*)d_in[11];
  const float* bc = (const float*)d_in[12];
  float* Hout = (float*)d_out;
  char* ws = (char*)d_ws;

  const size_t off_bias = (size_t)M_ * K_ * sizeof(_Float16);                // 160 KB
  const size_t off_S    = (off_bias + M_ * sizeof(float) + 255) & ~(size_t)255;
  const size_t off_P    = off_S + (size_t)N_ * O_ * sizeof(float);           // +64 MB
  const size_t needed   = off_P + (size_t)N_ * M_ * sizeof(float);           // +320 MB
  if (ws_size < needed) return;

  _Float16* Wh   = (_Float16*)ws;
  float* biasAll = (float*)(ws + off_bias);
  float* S       = (float*)(ws + off_S);
  float* P       = (float*)(ws + off_P);

  pack_kernel<<<dim3((M_ + 127) / 128), dim3(128), 0, stream>>>(
      wf, wi, wo, wc, bf, bi, bo, bc, Wh, biasAll);

  proj_gemm<<<dim3(N_ / 16), dim3(256), 0, stream>>>(x, Wh, biasAll, P);

  for (int d = 0; d <= 2 * (TGRID - 1); ++d) {
    int i0 = d > TGRID - 1 ? d - (TGRID - 1) : 0;
    int i1 = d < TGRID - 1 ? d : TGRID - 1;
    mdlstm_tile<<<dim3(i1 - i0 + 1), dim3(B_ * O_), 0, stream>>>(
        P, uf, ui, uo, uc, S, Hout, d, i0);
  }
}
